// PhysicsNetwork_v2_63943473102848
// MI455X (gfx1250) — matrix operand order repaired and checked by reading, not compiled, not run on hardware
//
#include <hip/hip_runtime.h>
#include <hip/hip_bf16.h>
#include <cstdint>

// ---------------------------------------------------------------------------
// CDNA5 (gfx1250, wave32) WMMA implementation of the interaction network.
// All GEMMs run through one templated wave32 WMMA kernel using
// v_wmma_f32_16x16x32_f16 (f32 activations converted to f16 in LDS, f32 accum).
// Weights are pre-transposed to f16 W^T[n][k] ONCE per launch, so the GEMM's
// B staging in the hot loop is pure b128 moves (no convert, no scatter).
// Wave tile 32x32 (4 WMMA sharing 2 A- and 2 B-fragments), block tile 64x128.
// FULL=true instantiations (all hot GEMMs: M%64==0, N%128==0, K%32==0) have
// zero bound checks. Guarded variants exist only for the K=4 / N=1 launches.
// The 61440-row pairwise branch gathers concat(state_i, state_j) rows directly
// in the GEMM A-loader (no 251 MB pairs tensor), chunked to stay L2-resident
// (192 MB L2, 23.3 TB/s HBM).
// ---------------------------------------------------------------------------

typedef __attribute__((ext_vector_type(16))) _Float16 v16h;
typedef __attribute__((ext_vector_type(8)))  _Float16 v8h;
typedef __attribute__((ext_vector_type(8)))  float    v8f;

#define ACT_NONE 0
#define ACT_ELU  1
#define ACT_SIG  2
#define A_SINGLE 0
#define A_CONCAT 1
#define A_PAIR   2

#define BM 64
#define BN 128
#define BK 32
#define LDSTR 40   // padded LDS stride in halves (80B rows, 16B aligned)

template<int ACT, int MODE, bool FULL>
__global__ void __launch_bounds__(256)
gemm_wmma(const float* __restrict__ A0, int lda0, int w0,
          const float* __restrict__ A1, int lda1, int w1,
          const _Float16* __restrict__ WT,   // f16 W^T, [N][Ktot]
          const float* __restrict__ bias,
          float* __restrict__ C, int ldc,
          int M, int N, int Ktot, int r0)
{
  __shared__ __align__(16) _Float16 As[BM][LDSTR];   // [m][k]
  __shared__ __align__(16) _Float16 Bs[BN][LDSTR];   // [n][k]

  const int tid  = threadIdx.x;
  const int lane = tid & 31;
  const int lr   = lane & 15;
  const int hi   = lane >> 4;          // lane half select
  const int wid  = tid >> 5;           // 8 waves
  const int wm   = (wid & 1) * 32;     // 2 waves along M -> 64
  const int wn   = (wid >> 1) * 32;    // 4 waves along N -> 128

  const int bm0 = blockIdx.y * BM;
  const int bn0 = blockIdx.x * BN;

  // ---- A staging coords: 64 rows x 32 k, 8 contiguous k per thread ----
  const int arow = tid >> 2;           // 0..63
  const int acol = (tid & 3) * 8;      // 0,8,16,24
  const int agm  = bm0 + arow;

  const float* aptr0 = A0;
  const float* aptr1 = A1;
  if (MODE == A_SINGLE) {
    aptr0 = A0 + (size_t)agm * lda0;
  } else if (MODE == A_CONCAT) {
    aptr0 = A0 + (size_t)agm * lda0;
    aptr1 = A1 + (size_t)agm * lda1;
  } else { // A_PAIR: row r -> (batch b, pair (i,j)); cols [0,w0) from row i,
           // cols [w0,w0+w1) from row j of state_enc (= A0)
    int r = r0 + agm;
    int t = r / 15;
    int m = r - t * 15;
    int b = t >> 4;
    int i = t & 15;
    int j = (m < i) ? m : (m + 1);     // skip-diagonal order of np.nonzero(~eye)
    aptr0 = A0 + (size_t)(b * 16 + i) * lda0;
    aptr1 = A0 + (size_t)(b * 16 + j) * lda0;
  }

  // ---- B staging coords: 128 n-rows x 32 k halves; 2 threads per n-row ----
  const int bn  = tid >> 1;            // 0..127
  const int bkh = (tid & 1) * 16;      // 0 or 16 (halves within k-slab)

  v8f acc[2][2] = {};

  for (int k0 = 0; k0 < Ktot; k0 += BK) {
    // ---- stage A tile (f32 -> f16): 2x b128 load, packed cvt, b128 store.
    // 8-element runs never straddle the concat boundary (w0 % 8 == 0).
    {
      const int gk0 = k0 + acol;
      v8h hv;
      if (FULL || (agm < M && gk0 + 8 <= Ktot)) {
        const float* src;
        if (MODE == A_SINGLE) src = aptr0 + gk0;
        else                  src = (gk0 < w0) ? (aptr0 + gk0)
                                               : (aptr1 + (gk0 - w0));
        const float4 x0 = ((const float4*)src)[0];
        const float4 x1 = ((const float4*)src)[1];
        hv[0] = (_Float16)x0.x; hv[1] = (_Float16)x0.y;
        hv[2] = (_Float16)x0.z; hv[3] = (_Float16)x0.w;
        hv[4] = (_Float16)x1.x; hv[5] = (_Float16)x1.y;
        hv[6] = (_Float16)x1.z; hv[7] = (_Float16)x1.w;
      } else {                          // cold path (K=4 launch, tails)
        #pragma unroll
        for (int e = 0; e < 8; ++e) {
          int gk = gk0 + e;
          float v = 0.f;
          if (agm < M && gk < Ktot) {
            if (MODE == A_SINGLE) v = aptr0[gk];
            else                  v = (gk < w0) ? aptr0[gk] : aptr1[gk - w0];
          }
          hv[e] = (_Float16)v;
        }
      }
      *(v8h*)&As[arow][acol] = hv;
    }
    // ---- stage B tile from pre-transposed f16 W^T[n][k]: pure b128 moves.
    {
      const int gn = bn0 + bn;
      v8h h0, h1;
      if (FULL) {
        const _Float16* src = &WT[(size_t)gn * Ktot + k0 + bkh];
        h0 = ((const v8h*)src)[0];
        h1 = ((const v8h*)src)[1];
        if (k0 + BK < Ktot)
          __builtin_prefetch(&WT[(size_t)gn * Ktot + k0 + BK + bkh], 0, 1);
      } else {                          // cold path (N=1 launches, K tails)
        #pragma unroll
        for (int e = 0; e < 16; ++e) {
          int gk = k0 + bkh + e;
          _Float16 v = (_Float16)0.f;
          if (gk < Ktot && gn < N) v = WT[(size_t)gn * Ktot + gk];
          if (e < 8) h0[e] = v; else h1[e - 8] = v;
        }
      }
      *(v8h*)&Bs[bn][bkh]     = h0;
      *(v8h*)&Bs[bn][bkh + 8] = h1;
    }
    __syncthreads();

    // ---- fragments. A (16x32 f16): lanes 0-15 hold K 0..15,
    // lanes 16-31 hold K 16..31; row M = (lane&15).
    v16h af[2];
    #pragma unroll
    for (int mt = 0; mt < 2; ++mt) {
      const _Float16* ar = &As[wm + mt * 16 + lr][0];
      const int kb0 = hi ? 16 : 0;
      const int kb1 = hi ? 24 : 8;
      #pragma unroll
      for (int v = 0; v < 4; ++v) {
        af[mt][2*v]     = ar[kb0 + 2*v];
        af[mt][2*v + 1] = ar[kb0 + 2*v + 1];
        af[mt][8 + 2*v]     = ar[kb1 + 2*v];
        af[mt][8 + 2*v + 1] = ar[kb1 + 2*v + 1];
      }
    }
    // B (32x16): lanes 0-15 K=0..15, lanes 16-31 K=16..31; col N = (lane&15).
    v16h bf[2];
    #pragma unroll
    for (int nt = 0; nt < 2; ++nt) {
      const _Float16* br = &Bs[wn + nt * 16 + lr][0];
      const int kb = hi ? 16 : 0;
      #pragma unroll
      for (int v = 0; v < 8; ++v) {
        bf[nt][2*v]     = br[kb + 2*v];
        bf[nt][2*v + 1] = br[kb + 2*v + 1];
      }
    }
    // 4 WMMA per wave per K-step; fragments reused across the 2x2 tile
    #pragma unroll
    for (int mt = 0; mt < 2; ++mt)
      #pragma unroll
      for (int nt = 0; nt < 2; ++nt)
        acc[mt][nt] = __builtin_amdgcn_wmma_f32_16x16x32_f16(
            false, af[mt], false, bf[nt], (short)0, acc[mt][nt], false, false);
    __syncthreads();
  }

  // ---- epilogue: C/D layout -> VGPR g: M = (hi?8:0)+g, N = lane%16
  #pragma unroll
  for (int mt = 0; mt < 2; ++mt) {
    #pragma unroll
    for (int nt = 0; nt < 2; ++nt) {
      #pragma unroll
      for (int g = 0; g < 8; ++g) {
        int om = bm0 + wm + mt * 16 + (hi ? 8 : 0) + g;
        int on = bn0 + wn + nt * 16 + lr;
        if (FULL || (om < M && on < N)) {
          float xv = acc[mt][nt][g] + bias[on];
          if (ACT == ACT_ELU)      xv = (xv > 0.f) ? xv : (__expf(xv) - 1.f);
          else if (ACT == ACT_SIG) xv = 1.f / (1.f + __expf(-xv));
          C[(size_t)om * ldc + on] = xv;
        }
      }
    }
  }
}

// one-time weight prep: WT[n][k] (f16) = W[k][n] (f32); coalesced stores
__global__ void transpose_w_kernel(const float* __restrict__ W,
                                   _Float16* __restrict__ WT, int K, int N) {
  int idx = blockIdx.x * blockDim.x + threadIdx.x;
  if (idx >= K * N) return;
  int n = idx / K, k = idx - n * K;
  WT[(size_t)n * K + k] = (_Float16)W[(size_t)k * N + n];
}

// state_enc = eff * att  (attention broadcast over feature dim)
__global__ void mul_bcast_kernel(const float* __restrict__ eff,
                                 const float* __restrict__ att,
                                 float* __restrict__ out, int M, int N) {
  int idx = blockIdx.x * blockDim.x + threadIdx.x;
  if (idx < M * N) out[idx] = eff[idx] * att[idx / N];
}

// total_effect[t0+tt][c] = sum_{m<15} e[tt*15+m][c] * a[tt*15+m]
__global__ void reduce15_kernel(const float* __restrict__ e,
                                const float* __restrict__ a,
                                float* __restrict__ out,
                                int T, int N, int t0) {
  int idx = blockIdx.x * blockDim.x + threadIdx.x;
  if (idx >= T * N) return;
  int tt = idx / N, c = idx - tt * N;
  float s = 0.f;
  #pragma unroll
  for (int m = 0; m < 15; ++m)
    s += e[(size_t)(tt * 15 + m) * N + c] * a[tt * 15 + m];
  out[(size_t)(t0 + tt) * N + c] = s;
}

// ---------------------------------------------------------------------------
// Host side
// ---------------------------------------------------------------------------

struct GArgs {
  const float* A0; int lda0, w0;
  const float* A1; int lda1, w1;
  const _Float16* WT; const float* b;
  float* C; int ldc;
  int M, N, K, r0;
};

static void launch_gemm(hipStream_t s, int act, int mode, const GArgs& g) {
  dim3 grid((g.N + BN - 1) / BN, (g.M + BM - 1) / BM), blk(256);
  const bool full = (g.M % BM == 0) && (g.N % BN == 0) && (g.K % BK == 0);
  #define GL(A, Md, Fl) gemm_wmma<A, Md, Fl><<<grid, blk, 0, s>>>( \
      g.A0, g.lda0, g.w0, g.A1, g.lda1, g.w1, g.WT, g.b, g.C, g.ldc, g.M, g.N, g.K, g.r0)
  if (mode == A_SINGLE) {
    if (act == ACT_ELU) {
      if (full) GL(ACT_ELU, A_SINGLE, true); else GL(ACT_ELU, A_SINGLE, false);
    } else if (act == ACT_SIG) {
      GL(ACT_SIG, A_SINGLE, false);            // only used with N=1
    } else {
      GL(ACT_NONE, A_SINGLE, true);            // only used fully aligned
    }
  } else if (mode == A_CONCAT) {
    GL(ACT_ELU, A_CONCAT, true);               // always 4096 x 512 x 1024
  } else {
    GL(ACT_ELU, A_PAIR, true);                 // always 1920 x 1024 x 1024
  }
  #undef GL
}

extern "C" void kernel_launch(void* const* d_in, const int* in_sizes, int n_in,
                              void* d_out, int out_size, void* d_ws, size_t ws_size,
                              hipStream_t stream) {
  (void)in_sizes; (void)out_size; (void)ws_size;
  if (n_in < 47) return;
  const float* x    = (const float*)d_in[0];   // sampled_state (4096, 512)
  const float* acts = (const float*)d_in[1];   // actions       (4096, 4)
  // d_in[2] = K scalar (16), fixed by setup
  const float* P[44];
  for (int i = 0; i < 44; ++i) P[i] = (const float*)d_in[3 + i];
  // params in setup_inputs() insertion order, each module {W1,b1,W2,b2}:
  // 0 inertia, 1 action_encoder, 2 action_effect, 3 action_attention,
  // 4 pairwise, 5 inter_effect, 6 inter_attention, 7 final_merge,
  // 8 det_out, 9 lam1, 10 lam2
  auto W1 = [&](int m) { return P[4 * m + 0]; };
  auto B1 = [&](int m) { return P[4 * m + 1]; };
  auto W2 = [&](int m) { return P[4 * m + 2]; };
  auto B2 = [&](int m) { return P[4 * m + 3]; };

  const int R = 4096, F = 512, AENC = 512, DET = 256, STO = 256;
  const int CH_T = 128, CH_R = CH_T * 15, NCH = R / CH_T;  // 32 chunks of 1920 rows

  float* ws = (float*)d_ws;
  size_t off = 0;
  auto give = [&](size_t n) { float* p = ws + off; off += n; return p; };
  float* buf_h  = give((size_t)R * F);       // shared hidden temp
  float* senc   = give((size_t)R * F);       // state_enc_flat
  float* aenc   = give((size_t)R * F);       // action_enc
  float* eff    = give((size_t)R * F);
  float* att    = give(R);
  float* stenc  = give((size_t)R * F);       // state_enc = eff*att
  float* toteff = give((size_t)R * AENC);
  float* agg    = give((size_t)R * F);
  float* pw_h   = give((size_t)CH_R * 1024); // chunk: pairwise hidden / e_h / a_h
  float* pw     = give((size_t)CH_R * F);
  float* e_buf  = give((size_t)CH_R * AENC);
  float* a_buf  = give(CH_R);

  // f16 region for pre-transposed weights (16B aligned; all sizes % 8 == 0)
  off = (off + 3) & ~(size_t)3;
  _Float16* hbase = (_Float16*)(ws + off);
  size_t hoff = 0;
  auto giveh = [&](size_t n) { _Float16* p = hbase + hoff; hoff += n; return p; };

  // module dims (din, dh, dout)
  const int din_[11]  = {512, 4, 1024, 1024, 1024, 512, 512, 1024, 512, 512, 512};
  const int dh_[11]   = {512, 512, 512, 512, 1024, 512, 512, 512, 512, 512, 512};
  const int dout_[11] = {512, 512, 512, 1, 512, 512, 1, 512, 256, 256, 256};
  const _Float16* WT1[11];
  const _Float16* WT2[11];
  for (int m = 0; m < 11; ++m) {
    int k1 = din_[m], n1 = dh_[m];
    _Float16* t1 = giveh((size_t)k1 * n1);
    transpose_w_kernel<<<((size_t)k1 * n1 + 255) / 256, 256, 0, stream>>>(
        W1(m), t1, k1, n1);
    WT1[m] = t1;
    int k2 = dh_[m], n2 = dout_[m];
    _Float16* t2 = giveh((size_t)k2 * n2);
    transpose_w_kernel<<<((size_t)k2 * n2 + 255) / 256, 256, 0, stream>>>(
        W2(m), t2, k2, n2);
    WT2[m] = t2;
  }

  float* out_det = (float*)d_out;
  float* out_l1  = out_det + (size_t)R * DET;
  float* out_l2  = out_l1 + (size_t)R * STO;

  auto S = [&](int act, const float* A, int lda, const _Float16* WT,
               const float* b, float* C, int ldc, int M, int N, int K) {
    GArgs gg{A, lda, K, nullptr, 0, 0, WT, b, C, ldc, M, N, K, 0};
    launch_gemm(stream, act, A_SINGLE, gg);
  };
  auto CC = [&](const float* A0, const float* A1, const _Float16* WT,
                const float* b, float* C, int M, int N) {
    GArgs gg{A0, F, F, A1, F, F, WT, b, C, N, M, N, 2 * F, 0};
    launch_gemm(stream, ACT_ELU, A_CONCAT, gg);
  };
  auto PAIR = [&](const float* SE, const _Float16* WT, const float* b,
                  float* C, int M, int N, int r0) {
    GArgs gg{SE, F, F, SE, F, F, WT, b, C, N, M, N, 2 * F, r0};
    launch_gemm(stream, ACT_ELU, A_PAIR, gg);
  };

  // ---- phase 1: per-entity encoders (4096 rows) ----
  S(ACT_ELU, x, F, WT1[0], B1(0), buf_h, F, R, F, F);          // inertia h
  S(ACT_ELU, buf_h, F, WT2[0], B2(0), senc, F, R, F, F);       // state_enc_flat
  S(ACT_ELU, acts, 4, WT1[1], B1(1), buf_h, F, R, F, 4);       // action_encoder h
  S(ACT_ELU, buf_h, F, WT2[1], B2(1), aenc, F, R, F, F);       // action_enc
  CC(senc, aenc, WT1[2], B1(2), buf_h, R, F);                  // action_effect h
  S(ACT_ELU, buf_h, F, WT2[2], B2(2), eff, F, R, F, F);
  CC(senc, aenc, WT1[3], B1(3), buf_h, R, F);                  // action_attention h
  S(ACT_SIG, buf_h, F, WT2[3], B2(3), att, 1, R, 1, F);
  mul_bcast_kernel<<<(R * F + 255) / 256, 256, 0, stream>>>(eff, att, stenc, R, F);

  // ---- phase 2: pairwise interactions, gather fused into GEMM, chunked ----
  for (int c = 0; c < NCH; ++c) {
    int r0 = c * CH_R;
    PAIR(stenc, WT1[4], B1(4), pw_h, CH_R, 1024, r0);                // pairwise h
    S(ACT_ELU, pw_h, 1024, WT2[4], B2(4), pw, F, CH_R, F, 1024);     // pw
    S(ACT_ELU, pw, F, WT1[5], B1(5), pw_h, F, CH_R, F, F);           // e_h
    S(ACT_ELU, pw_h, F, WT2[5], B2(5), e_buf, AENC, CH_R, AENC, F);  // e
    S(ACT_ELU, pw, F, WT1[6], B1(6), pw_h, F, CH_R, F, F);           // a_h
    S(ACT_SIG, pw_h, F, WT2[6], B2(6), a_buf, 1, CH_R, 1, F);        // a
    reduce15_kernel<<<(CH_T * AENC + 255) / 256, 256, 0, stream>>>(
        e_buf, a_buf, toteff, CH_T, AENC, c * CH_T);                 // total_effect
  }

  // ---- phase 3: merge + heads ----
  CC(stenc, toteff, WT1[7], B1(7), buf_h, R, F);               // final_merge h
  S(ACT_ELU, buf_h, F, WT2[7], B2(7), agg, F, R, F, F);        // agg
  S(ACT_ELU, agg, F, WT1[8], B1(8), buf_h, F, R, F, F);        // det_out h
  S(ACT_ELU, buf_h, F, WT2[8], B2(8), out_det, DET, R, DET, F);// deter_state
  S(ACT_ELU, agg, F, WT1[9], B1(9), buf_h, F, R, F, F);        // lam1 h
  S(ACT_NONE, buf_h, F, WT2[9], B2(9), out_l1, STO, R, STO, F);// lambdas1
  S(ACT_ELU, agg, F, WT1[10], B1(10), buf_h, F, R, F, F);      // lam2 h
  S(ACT_NONE, buf_h, F, WT2[10], B2(10), out_l2, STO, R, STO, F); // lambdas2
}
